// MultiHeadAttn_62981400428865
// MI455X (gfx1250) — matrix operand order repaired and checked by reading, not compiled, not run on hardware
//
#include <hip/hip_runtime.h>
#include <hip/hip_bf16.h>

// Problem constants (from reference)
#define B_    2
#define NQ_   2048
#define SK_   2048
#define ISZ_  1024
#define HSZ_  1024
#define OSZ_  1024
#define H_    16
#define ADIM_ 64
#define IEPS_ 1e-32f

typedef __bf16 bf16_t;
typedef __attribute__((ext_vector_type(16))) __bf16 v16bf;
typedef __attribute__((ext_vector_type(8)))  __bf16 v8bf;
typedef __attribute__((ext_vector_type(8)))  float  v8f;

// ---------------------------------------------------------------------------
// WMMA helper: D = A(16x32 bf16) x B(32x16 bf16) + C(16x16 f32)
// ---------------------------------------------------------------------------
__device__ __forceinline__ v8f wmma_bf16(v16bf a, v16bf b, v8f c) {
  return __builtin_amdgcn_wmma_f32_16x16x32_bf16(
      /*neg_a=*/false, a, /*neg_b=*/false, b,
      /*c_mod=*/(short)0, c, /*reuse_a=*/false, /*reuse_b=*/false);
}

// A-matrix fragment loader (16x32 A, contiguous k order matched by the B loads, not the ISA table order):
//   lane L (row m = L&15, half = L>>4):
//     elems 0..15 = A[m][k0 + half*16 .. +15]
// rowptr must point at A[m] + k0 (16B aligned for k0 multiple of 8).
__device__ __forceinline__ v16bf load_a_frag(const bf16_t* rowptr, int half) {
  v8bf lo = *reinterpret_cast<const v8bf*>(rowptr + half * 16);
  v8bf hi = *reinterpret_cast<const v8bf*>(rowptr + half * 16 + 8);
  v16bf a;
#pragma unroll
  for (int i = 0; i < 8; ++i) { a[i] = lo[i]; a[8 + i] = hi[i]; }
  return a;
}

// ---------------------------------------------------------------------------
// fp32 -> bf16 elementwise conversion
// ---------------------------------------------------------------------------
__global__ void f32_to_bf16_kernel(const float* __restrict__ in,
                                   bf16_t* __restrict__ out, int n) {
  int i = blockIdx.x * blockDim.x + threadIdx.x;
  int stride = gridDim.x * blockDim.x;
  for (; i < n; i += stride) out[i] = (bf16_t)in[i];
}

// ---------------------------------------------------------------------------
// Generic GEMM: C[M][N] = A[M][K] @ W[N][K]^T + bias[N]
// A, W in bf16 row-major. Block = 128 threads (4 waves). Each wave owns a
// 32(M) x 64(N) tile (2 M-subtiles x 4 N-tiles, 8 WMMAs per k-step, B
// fragments reused across both M-subtiles). Block covers 128x64.
// Grid = (N/64, M/128).
// Epilogue modes:
//   0: bf16 out, head-major  out[((b*H + h)*LEN + q)*64 + d]   (Q, K)
//   1: bf16 out, head-transposed out[((b*H + h)*64 + d)*LEN + q]  (V^T)
//   2: f32 plain out[m*N + n]                                   (O proj)
// ---------------------------------------------------------------------------
__global__ void __launch_bounds__(128)
gemm_bf16_wmma(const bf16_t* __restrict__ A, const bf16_t* __restrict__ W,
               const float* __restrict__ bias, void* __restrict__ out,
               int M, int N, int K, int mode, int LEN) {
  const int lane = threadIdx.x & 31;
  const int wave = threadIdx.x >> 5;
  const int half = lane >> 4;
  const int l16  = lane & 15;
  const int m0 = blockIdx.y * 128 + wave * 32;
  const int n0 = blockIdx.x * 64;

  v8f acc[2][4] = {};  // [M-subtile][N-tile] of 16x16 f32

  const bf16_t* arow0 = A + (size_t)(m0 + l16) * K;
  const bf16_t* arow1 = A + (size_t)(m0 + 16 + l16) * K;
  for (int k0 = 0; k0 < K; k0 += 32) {
    v16bf a0 = load_a_frag(arow0 + k0, half);
    v16bf a1 = load_a_frag(arow1 + k0, half);
#pragma unroll
    for (int t = 0; t < 4; ++t) {
      // B-matrix col n = W row n (since C = A @ W^T): lane l16 holds column
      // n0+t*16+l16; half selects K 0..15 vs 16..31 of this 32-step.
      const bf16_t* wrow = W + (size_t)(n0 + t * 16 + l16) * K + k0 + half * 16;
      v16bf b = *reinterpret_cast<const v16bf*>(wrow);
      acc[0][t] = wmma_bf16(a0, b, acc[0][t]);
      acc[1][t] = wmma_bf16(a1, b, acc[1][t]);
    }
  }

  // C layout: element e of acc[ms][t] at lane L is
  //   (m = m0 + ms*16 + e + half*8, n = n0 + t*16 + l16)
#pragma unroll
  for (int ms = 0; ms < 2; ++ms) {
#pragma unroll
    for (int t = 0; t < 4; ++t) {
      const int n = n0 + t * 16 + l16;
      const float bv = bias[n];
#pragma unroll
      for (int e = 0; e < 8; ++e) {
        const int m = m0 + ms * 16 + e + half * 8;
        const float v = acc[ms][t][e] + bv;
        if (mode == 2) {
          reinterpret_cast<float*>(out)[(size_t)m * N + n] = v;
        } else {
          const int b = m / LEN, q = m % LEN;
          const int h = n >> 6, d = n & 63;
          bf16_t* o = reinterpret_cast<bf16_t*>(out);
          if (mode == 0)
            o[(((size_t)(b * H_ + h) * LEN + q) << 6) + d] = (bf16_t)v;
          else
            o[((size_t)(b * H_ + h) * ADIM_ + d) * LEN + q] = (bf16_t)v;
        }
      }
    }
  }
}

// ---------------------------------------------------------------------------
// Fused SparseNormer attention. One wave = 32 query rows of one (b,h).
// Streams over 32-key chunks: score WMMAs (K=adim=64 -> 2 per 16-key tile
// per M-subtile), t = relu(score/8 + nbias)^2 (masked -> 0), accumulate
// rowsum + t@V. Every K-row/V-row fetch feeds two WMMAs (both M-subtiles).
// t tiles staged through LDS to convert C-layout -> A-layout for the attend
// WMMA. Single pass (normalizer is a plain sum, no running max needed).
// Grid = (NQ/128, B*H), block = 128 (4 waves).
// ---------------------------------------------------------------------------
__global__ void __launch_bounds__(128)
attn_sparse_normer(const bf16_t* __restrict__ qh, const bf16_t* __restrict__ kh,
                   const bf16_t* __restrict__ vt, const int* __restrict__ mask,
                   const float* __restrict__ nbias_p, bf16_t* __restrict__ oh) {
  __shared__ __align__(64) bf16_t tile[4][32][32];  // per-wave [m][key] probs

  const int lane = threadIdx.x & 31;
  const int wave = threadIdx.x >> 5;
  const int half = lane >> 4;
  const int l16  = lane & 15;
  const int bh = blockIdx.y;          // b*H + h
  const int b  = bh >> 4;             // H_ == 16
  const int h  = bh & 15;
  const int qbase = blockIdx.x * 128 + wave * 32;
  const float nbias = nbias_p[0];
  const float scale = 0.125f;         // 1/sqrt(64)

  // Preload the 32x64 Q tile: [M-subtile][K-step] A-fragments.
  v16bf aq[2][2];
#pragma unroll
  for (int ms = 0; ms < 2; ++ms) {
    const bf16_t* qrow = qh + (((size_t)bh * NQ_ + qbase + ms * 16 + l16) << 6);
    aq[ms][0] = load_a_frag(qrow, half);
    aq[ms][1] = load_a_frag(qrow + 32, half);
  }

  v8f acc[2][4] = {};  // un-normalized output, [M-subtile][adim tile]
  v8f rs[2] = {};      // per-(m, n-lane) partial rowsums

  const bf16_t* kbase = kh + ((size_t)bh * SK_ << 6);
  const bf16_t* vbase = vt + (size_t)bh * ADIM_ * SK_;
  const int* mbase = mask + ((size_t)b * NQ_ + qbase) * SK_;

  for (int kt2 = 0; kt2 < SK_ / 32; ++kt2) {
#pragma unroll
    for (int sub = 0; sub < 2; ++sub) {
      const int key0 = kt2 * 32 + sub * 16;
      // B = K^T tile: column n = K row (key0+l16); contiguous 16 bf16.
      const bf16_t* krow = kbase + ((size_t)(key0 + l16) << 6) + half * 16;
      v16bf b0 = *reinterpret_cast<const v16bf*>(krow);
      v16bf b1 = *reinterpret_cast<const v16bf*>(krow + 32);
      const int key = key0 + l16;
#pragma unroll
      for (int ms = 0; ms < 2; ++ms) {
        v8f s = {};
        s = wmma_bf16(aq[ms][0], b0, s);
        s = wmma_bf16(aq[ms][1], b1, s);
#pragma unroll
        for (int e = 0; e < 8; ++e) {
          const int m = ms * 16 + e + half * 8;           // row in 32-row tile
          const int mk = mbase[(size_t)m * SK_ + key];    // true -> masked
          float sv = s[e] * scale + nbias;
          sv = (mk != 0 || sv <= 0.0f) ? 0.0f : sv;
          const float t = sv * sv;                        // relu(x)^2
          rs[ms][e] += t;
          tile[wave][m][sub * 16 + l16] = (bf16_t)t;      // C-layout -> LDS
        }
      }
    }
    // Re-read 32x32 prob tile in A-layout (same-wave LDS ops are in-order).
    v16bf at0 = load_a_frag(&tile[wave][l16][0], half);
    v16bf at1 = load_a_frag(&tile[wave][16 + l16][0], half);
#pragma unroll
    for (int nt = 0; nt < 4; ++nt) {
      // B column d = V^T row (nt*16+l16); contiguous 16 bf16 at this K-chunk.
      const bf16_t* vrow =
          vbase + (size_t)(nt * 16 + l16) * SK_ + kt2 * 32 + half * 16;
      v16bf bv = *reinterpret_cast<const v16bf*>(vrow);
      acc[0][nt] = wmma_bf16(at0, bv, acc[0][nt]);
      acc[1][nt] = wmma_bf16(at1, bv, acc[1][nt]);
    }
  }

  // Rowsum reduction: sum over the 16 N-lanes of each half (xor 8/4/2/1
  // never crosses the lane-half boundary). Result replicated per lane.
#pragma unroll
  for (int ms = 0; ms < 2; ++ms) {
#pragma unroll
    for (int e = 0; e < 8; ++e) {
      float r = rs[ms][e];
#pragma unroll
      for (int off = 8; off >= 1; off >>= 1) r += __shfl_xor(r, off, 32);
      rs[ms][e] = r;
    }
  }

  // Normalize and store head-major bf16: oh[b][q][h*64 + d].
#pragma unroll
  for (int ms = 0; ms < 2; ++ms) {
#pragma unroll
    for (int nt = 0; nt < 4; ++nt) {
#pragma unroll
      for (int e = 0; e < 8; ++e) {
        const int q = qbase + ms * 16 + e + half * 8;
        const int d = (h << 6) + nt * 16 + l16;
        const float v = acc[ms][nt][e] / (rs[ms][e] + IEPS_);
        oh[((size_t)b * NQ_ + q) * HSZ_ + d] = (bf16_t)v;
      }
    }
  }
}

// ---------------------------------------------------------------------------
// Host-side orchestration
// ---------------------------------------------------------------------------
extern "C" void kernel_launch(void* const* d_in, const int* in_sizes, int n_in,
                              void* d_out, int out_size, void* d_ws, size_t ws_size,
                              hipStream_t stream) {
  (void)in_sizes; (void)n_in; (void)out_size; (void)ws_size;

  const float* iQ  = (const float*)d_in[0];
  const float* iK  = (const float*)d_in[1];
  const float* iV  = (const float*)d_in[2];
  const int*   msk = (const int*)d_in[3];
  const float* Wq  = (const float*)d_in[4];
  const float* bq  = (const float*)d_in[5];
  const float* Wk  = (const float*)d_in[6];
  const float* bk  = (const float*)d_in[7];
  const float* Wv  = (const float*)d_in[8];
  const float* bv  = (const float*)d_in[9];
  const float* Wo  = (const float*)d_in[10];
  const float* bo  = (const float*)d_in[11];
  const float* nb  = (const float*)d_in[12];

  const size_t nAct = (size_t)B_ * NQ_ * ISZ_;   // 4.19M (NQ==SK)
  const size_t nW   = (size_t)HSZ_ * ISZ_;       // 1.05M
  const size_t nHd  = (size_t)B_ * H_ * NQ_ * ADIM_;  // 4.19M

  char* ws = (char*)d_ws;
  size_t off = 0;
  auto take = [&](size_t elems) -> bf16_t* {
    bf16_t* p = (bf16_t*)(ws + off);
    off = (off + elems * sizeof(bf16_t) + 255) & ~(size_t)255;
    return p;
  };
  bf16_t* iQb = take(nAct);
  bf16_t* iKb = take(nAct);
  bf16_t* iVb = take(nAct);
  bf16_t* Wqb = take(nW);
  bf16_t* Wkb = take(nW);
  bf16_t* Wvb = take(nW);
  bf16_t* Wob = take(nW);
  bf16_t* qh  = take(nHd);
  bf16_t* kh  = take(nHd);
  bf16_t* vt  = take(nHd);
  bf16_t* oh  = take((size_t)B_ * NQ_ * HSZ_);

  // 1) fp32 -> bf16 conversions
  const int cb = 256;
  auto cgrid = [&](size_t n) { size_t g = (n + cb - 1) / cb; return (int)(g > 4096 ? 4096 : g); };
  f32_to_bf16_kernel<<<cgrid(nAct), cb, 0, stream>>>(iQ, iQb, (int)nAct);
  f32_to_bf16_kernel<<<cgrid(nAct), cb, 0, stream>>>(iK, iKb, (int)nAct);
  f32_to_bf16_kernel<<<cgrid(nAct), cb, 0, stream>>>(iV, iVb, (int)nAct);
  f32_to_bf16_kernel<<<cgrid(nW),   cb, 0, stream>>>(Wq, Wqb, (int)nW);
  f32_to_bf16_kernel<<<cgrid(nW),   cb, 0, stream>>>(Wk, Wkb, (int)nW);
  f32_to_bf16_kernel<<<cgrid(nW),   cb, 0, stream>>>(Wv, Wvb, (int)nW);
  f32_to_bf16_kernel<<<cgrid(nW),   cb, 0, stream>>>(Wo, Wob, (int)nW);

  // 2) QKV projections (M = B*len, N = HSZ, K = ISZ)
  const int M = B_ * NQ_;
  dim3 gblk(128);
  dim3 ggrd(HSZ_ / 64, M / 128);
  gemm_bf16_wmma<<<ggrd, gblk, 0, stream>>>(iQb, Wqb, bq, qh, M, HSZ_, ISZ_, 0, NQ_);
  gemm_bf16_wmma<<<ggrd, gblk, 0, stream>>>(iKb, Wkb, bk, kh, M, HSZ_, ISZ_, 0, SK_);
  gemm_bf16_wmma<<<ggrd, gblk, 0, stream>>>(iVb, Wvb, bv, vt, M, HSZ_, ISZ_, 1, SK_);

  // 3) fused attention with SparseNormer
  dim3 agrd(NQ_ / 128, B_ * H_);
  attn_sparse_normer<<<agrd, gblk, 0, stream>>>(qh, kh, vt, msk, nb, oh);

  // 4) output projection -> fp32 d_out
  dim3 ogrd(OSZ_ / 64, M / 128);
  gemm_bf16_wmma<<<ogrd, gblk, 0, stream>>>(oh, Wob, bo, d_out, M, OSZ_, HSZ_, 2, NQ_);
}
